// MultiScaleDeformableAttention_37452114821812
// MI455X (gfx1250) — matrix operand order repaired and checked by reading, not compiled, not run on hardware
//
#include <hip/hip_runtime.h>

// ---------------------------------------------------------------------------
// MultiScaleDeformableAttention for MI455X (gfx1250, wave32, WMMA)
//   4 GEMMs via v_wmma_f32_16x16x32_f16 (f16 operands, f32 accumulate)
//   + coalesced bilinear-gather kernel (wave = head, lane = channel)
// Operands are pre-swizzled into the exact WMMA per-lane fragment layout
// (CDNA5 ISA 7.12.2) so the GEMM inner loop is b128 loads + wmma only.
// ---------------------------------------------------------------------------

typedef __attribute__((ext_vector_type(16))) _Float16 v16h;
typedef __attribute__((ext_vector_type(8)))  float    v8f;

#define BB      4
#define LQ      8400
#define LVTOT   8500
#define EE      256
#define NHEADS  8
#define HDIM    32

// ---- A-operand swizzle: fp32 [M][256] -> f16 WMMA A-fragments -------------
// A 16x32 f16 fragment (per ISA): lanes 0-15 M=lane, halves 0..7 = K 0..7,
// halves 8..15 = K 16..23; lanes 16-31 M=lane-16, halves = K 8..15 / 24..31.
__global__ void swizzleA_kernel(const float* __restrict__ A,
                                _Float16* __restrict__ out, int M) {
  int t = blockIdx.x * blockDim.x + threadIdx.x;
  if (t >= M * 256) return;
  int row = t >> 8, k = t & 255;
  int mtile = row >> 4, kb = k >> 5, kr = k & 31;
  int lane = (row & 15) + (((kr & 15) >> 3) << 4);
  int j    = (kr & 7) + ((kr >> 4) << 3);
  out[(((size_t)(mtile * 8 + kb) * 32 + lane) << 4) + j] = (_Float16)A[t];
}

// ---- B-operand swizzle: fp32 [K=256][N] -> f16 WMMA B-fragments -----------
// B 32x16 f16 fragment: lanes 0-15 N=lane, halves 0..7 = K 0..7,
// halves 8..15 = K 16..23; lanes 16-31 N=lane-16, halves = K 8..15 / 24..31.
__global__ void swizzleB_kernel(const float* __restrict__ W,
                                _Float16* __restrict__ out, int N) {
  int t = blockIdx.x * blockDim.x + threadIdx.x;
  if (t >= 256 * N) return;
  int k = t / N, n = t - k * N;
  int ntile = n >> 4, kb = k >> 5, kr = k & 31;
  int lane = (n & 15) + (((kr & 15) >> 3) << 4);
  int j    = (kr & 7) + ((kr >> 4) << 3);
  out[(((size_t)(ntile * 8 + kb) * 32 + lane) << 4) + j] = (_Float16)W[t];
}

// ---- WMMA GEMM: C[M][N] = A[M][256] * W[256][N] + bias --------------------
// grid.x = M/16 (exact), block = 256 threads = 8 waves.
// Wave w owns ntiles [w*NTW, w*NTW+NTW); 8 K-steps of K=32.
template <typename OutT, int NTW>
__global__ void gemm_wmma_kernel(const _Float16* __restrict__ Afrag,
                                 const _Float16* __restrict__ Bfrag,
                                 const float* __restrict__ bias,
                                 OutT* __restrict__ C, int N) {
  const int mtile = blockIdx.x;
  const int wave  = threadIdx.x >> 5;
  const int lane  = threadIdx.x & 31;

  v8f acc[NTW];
#pragma unroll
  for (int i = 0; i < NTW; ++i)
#pragma unroll
    for (int r = 0; r < 8; ++r) acc[i][r] = 0.0f;

#pragma unroll
  for (int kb = 0; kb < 8; ++kb) {
    v16h a = *(const v16h*)(Afrag + (((size_t)mtile * 8 + kb) * 32 + lane) * 16);
#pragma unroll
    for (int i = 0; i < NTW; ++i) {
      const int ntile = wave * NTW + i;
      v16h b = *(const v16h*)(Bfrag + (((size_t)ntile * 8 + kb) * 32 + lane) * 16);
      acc[i] = __builtin_amdgcn_wmma_f32_16x16x32_f16(
          /*neg_a=*/false, a, /*neg_b=*/false, b,
          /*c_mod=*/(short)0, acc[i], /*reuse_a=*/false, /*reuse_b=*/false);
    }
  }

  const int colLocal = lane & 15;
  const int rowOff   = (lane >> 4) * 8;   // C/D layout: lanes 16-31 hold M=r+8
#pragma unroll
  for (int i = 0; i < NTW; ++i) {
    const int ntile = wave * NTW + i;
    const int col   = ntile * 16 + colLocal;
    const float bc  = bias[col];
#pragma unroll
    for (int r = 0; r < 8; ++r) {
      const int row = mtile * 16 + rowOff + r;
      C[(size_t)row * N + col] = (OutT)(acc[i][r] + bc);
    }
  }
}

// ---- softmax + bilinear sampling ------------------------------------------
// grid.x = B*LQ, block = 256: wave = head (8 waves), lane = channel d (32).
// Writes result directly in A-fragment-swizzled f16 for the final GEMM.
__global__ void msda_sample_kernel(const _Float16* __restrict__ vproj, // [B][LV][256]
                                   const _Float16* __restrict__ off,   // [B*LQ][256]
                                   const _Float16* __restrict__ awr,   // [B*LQ][128]
                                   const float* __restrict__ refp,     // [B][LQ][4][2]
                                   _Float16* __restrict__ outA) {
  const int qg = blockIdx.x;           // b*LQ + q
  const int b  = qg / LQ;
  const int h  = threadIdx.x >> 5;
  const int d  = threadIdx.x & 31;

  // per-head softmax over LEVELS*POINTS = 16 logits (wave-uniform)
  float wv[16];
  float mx = -1e30f;
  const _Float16* ap = awr + (size_t)qg * 128 + h * 16;
#pragma unroll
  for (int i = 0; i < 16; ++i) { float a = (float)ap[i]; wv[i] = a; mx = fmaxf(mx, a); }
  float s = 0.0f;
#pragma unroll
  for (int i = 0; i < 16; ++i) { float e = __expf(wv[i] - mx); wv[i] = e; s += e; }
  const float inv = 1.0f / s;

  const _Float16* op = off + (size_t)qg * 256 + h * 32;
  const _Float16* vb = vproj + (size_t)b * LVTOT * 256 + h * 32 + d;

  const int Hs[4] = {80, 40, 20, 10};
  const int Ws[4] = {80, 40, 20, 10};
  const int st[4] = {0, 6400, 8000, 8400};

  float acc = 0.0f;
#pragma unroll
  for (int l = 0; l < 4; ++l) {
    const float rx = refp[((size_t)qg * 4 + l) * 2 + 0];
    const float ry = refp[((size_t)qg * 4 + l) * 2 + 1];
    const int Wl = Ws[l], Hl = Hs[l];
    const _Float16* vl = vb + (size_t)st[l] * 256;
#pragma unroll
    for (int p = 0; p < 4; ++p) {
      const float ox = (float)op[(l * 4 + p) * 2 + 0];
      const float oy = (float)op[(l * 4 + p) * 2 + 1];
      const float w  = wv[l * 4 + p] * inv;
      // grid math collapses to: x = ref_x*W + off_x - 0.5 (same for y)
      const float x = rx * (float)Wl + ox - 0.5f;
      const float y = ry * (float)Hl + oy - 0.5f;
      const float xf = floorf(x), yf = floorf(y);
      const float fx = x - xf, fy = y - yf;
      const int ix = (int)xf, iy = (int)yf;
      float v00 = 0.f, v01 = 0.f, v10 = 0.f, v11 = 0.f;
      const bool xin0 = (ix >= 0) & (ix < Wl);
      const bool xin1 = (ix + 1 >= 0) & (ix + 1 < Wl);
      if ((iy >= 0) & (iy < Hl)) {              // wave-uniform branches
        const _Float16* r0 = vl + (size_t)(iy * Wl) * 256;
        if (xin0) v00 = (float)r0[(size_t)ix * 256];
        if (xin1) v01 = (float)r0[(size_t)(ix + 1) * 256];
      }
      if ((iy + 1 >= 0) & (iy + 1 < Hl)) {
        const _Float16* r1 = vl + (size_t)((iy + 1) * Wl) * 256;
        if (xin0) v10 = (float)r1[(size_t)ix * 256];
        if (xin1) v11 = (float)r1[(size_t)(ix + 1) * 256];
      }
      acc += w * ((1.f - fx) * (1.f - fy) * v00 + fx * (1.f - fy) * v01 +
                  (1.f - fx) * fy * v10 + fx * fy * v11);
    }
  }

  // store in A-fragment swizzle: M = qg, K = h*32 + d  (kr = d)
  const int mtile = qg >> 4;
  const int lane2 = (qg & 15) + (((d & 15) >> 3) << 4);
  const int j     = (d & 7) + ((d >> 4) << 3);
  outA[(((size_t)(mtile * 8 + h) * 32 + lane2) << 4) + j] = (_Float16)acc;
}

// ---------------------------------------------------------------------------
extern "C" void kernel_launch(void* const* d_in, const int* in_sizes, int n_in,
                              void* d_out, int out_size, void* d_ws, size_t ws_size,
                              hipStream_t stream) {
  (void)in_sizes; (void)n_in; (void)out_size; (void)ws_size;

  const float* query = (const float*)d_in[0];
  const float* refp  = (const float*)d_in[1];
  const float* value = (const float*)d_in[2];
  /* d_in[3] = spatial_shapes (compile-time constant here) */
  const float* Wv    = (const float*)d_in[4];
  const float* bv    = (const float*)d_in[5];
  const float* Woff  = (const float*)d_in[6];
  const float* boff  = (const float*)d_in[7];
  const float* Wattn = (const float*)d_in[8];
  const float* battn = (const float*)d_in[9];
  const float* Wo    = (const float*)d_in[10];
  const float* bo    = (const float*)d_in[11];
  float* out = (float*)d_out;

  const int MQ = BB * LQ;     // 33600 (divisible by 16)
  const int MV = BB * LVTOT;  // 34000 (divisible by 16)

  // workspace carve-out (256B aligned slices)
  char* ws = (char*)d_ws;
  size_t o = 0;
  auto carve = [&](size_t bytes) -> char* {
    char* p = ws + o;
    o += (bytes + 255) & ~(size_t)255;
    return p;
  };
  _Float16* qA     = (_Float16*)carve((size_t)MQ * 256 * 2);
  _Float16* vA     = (_Float16*)carve((size_t)MV * 256 * 2);
  _Float16* WvB    = (_Float16*)carve((size_t)256 * 256 * 2);
  _Float16* WoffB  = (_Float16*)carve((size_t)256 * 256 * 2);
  _Float16* WattnB = (_Float16*)carve((size_t)256 * 128 * 2);
  _Float16* WoB    = (_Float16*)carve((size_t)256 * 256 * 2);
  _Float16* vproj  = (_Float16*)carve((size_t)MV * 256 * 2);
  _Float16* offb   = (_Float16*)carve((size_t)MQ * 256 * 2);
  _Float16* awrb   = (_Float16*)carve((size_t)MQ * 128 * 2);
  _Float16* attnA  = (_Float16*)carve((size_t)MQ * 256 * 2);

  // 1) swizzle operands into WMMA fragment layouts
  swizzleA_kernel<<<(MQ * 256 + 255) / 256, 256, 0, stream>>>(query, qA, MQ);
  swizzleA_kernel<<<(MV * 256 + 255) / 256, 256, 0, stream>>>(value, vA, MV);
  swizzleB_kernel<<<(256 * 256 + 255) / 256, 256, 0, stream>>>(Wv, WvB, 256);
  swizzleB_kernel<<<(256 * 256 + 255) / 256, 256, 0, stream>>>(Woff, WoffB, 256);
  swizzleB_kernel<<<(256 * 128 + 255) / 256, 256, 0, stream>>>(Wattn, WattnB, 128);
  swizzleB_kernel<<<(256 * 256 + 255) / 256, 256, 0, stream>>>(Wo, WoB, 256);

  // 2) projections (WMMA)
  gemm_wmma_kernel<_Float16, 2><<<MV / 16, 256, 0, stream>>>(vA, WvB, bv, vproj, 256);
  gemm_wmma_kernel<_Float16, 2><<<MQ / 16, 256, 0, stream>>>(qA, WoffB, boff, offb, 256);
  gemm_wmma_kernel<_Float16, 1><<<MQ / 16, 256, 0, stream>>>(qA, WattnB, battn, awrb, 128);

  // 3) softmax + deformable bilinear sampling (writes swizzled A for final GEMM)
  msda_sample_kernel<<<MQ, 256, 0, stream>>>(vproj, offb, awrb, refp, attnA);

  // 4) output projection (WMMA, f32 out)
  gemm_wmma_kernel<float, 2><<<MQ / 16, 256, 0, stream>>>(attnA, WoB, bo, out, 256);
}
